// FramepairUpdate_59115929862890
// MI455X (gfx1250) — matrix operand order repaired and checked by reading, not compiled, not run on hardware
//
#include <hip/hip_runtime.h>
#include <stdint.h>

typedef __attribute__((ext_vector_type(16))) __bf16 v16bf;
typedef __attribute__((ext_vector_type(8)))  float  v8f;
typedef __attribute__((ext_vector_type(4)))  float  f4;

#define NRES 512
#define CZ   128
#define CF   384
#define CP   256
#define TM   16   // pair-rows per block

// ---------- A-matrix pack index (16-bit A 16x32 layout per ISA 7.12.2) ----------
__device__ __forceinline__ int pack_idx_a(int m, int kl) {
    int hf   = ((kl & 15) >= 8) ? 1 : 0;
    int lane = m + (hf << 4);
    int e    = (kl < 16) ? (kl & 7) : (8 + (kl & 7));
    return lane * 16 + e;   // within one k-tile block of 32 lanes * 16 elems
}

__device__ __forceinline__ v8f vzero8() {
    v8f r = {0.f, 0.f, 0.f, 0.f, 0.f, 0.f, 0.f, 0.f};
    return r;
}

__device__ __forceinline__ v8f wmma_bf16(v16bf a, v16bf b, v8f c) {
    return __builtin_amdgcn_wmma_f32_16x16x32_bf16(false, a, false, b, (short)0, c, false, false);
}

// ---------- weight prepack: fp32 KxCP -> bf16 B-fragments ----------
// [n_tile][k_tile][lane][16 elems]; B 32x16 16-bit layout:
// lanes 0-15 hold K=0..7,16..23 (col N=lane), lanes 16-31 hold K=8..15,24..31 (col N=lane-16)
__global__ void prepack_kernel(const float* __restrict__ W,
                               unsigned short* __restrict__ outp, int KT) {
    int idx   = blockIdx.x * blockDim.x + threadIdx.x;
    int total = 16 * KT * 512;
    if (idx >= total) return;
    int e    = idx & 15;
    int lane = (idx >> 4) & 31;
    int kt   = (idx >> 9) % KT;
    int nt   = idx / (KT << 9);
    int nl   = lane & 15;
    int hf   = lane >> 4;
    int k    = kt * 32 + 8 * hf + (e & 7) + ((e >> 3) << 4);
    int n    = nt * 16 + nl;
    __bf16 b = (__bf16)W[k * CP + n];
    outp[idx] = __builtin_bit_cast(unsigned short, b);
}

// ---------- stage 1: src = LN(rigids); l = src@W_l; m = src@W_m ----------
__global__ void lm_kernel(const float* __restrict__ rig,
                          const float* __restrict__ g, const float* __restrict__ b,
                          const float* __restrict__ Wl, const float* __restrict__ Wm,
                          float* __restrict__ lo, float* __restrict__ mo) {
    __shared__ float srow[CF];
    __shared__ float red[128];
    int n = blockIdx.x, tid = threadIdx.x;
    float x0 = rig[n * CF + tid];
    float x1 = rig[n * CF + tid + 128];
    float x2 = rig[n * CF + tid + 256];
    red[tid] = x0 + x1 + x2;
    __syncthreads();
    for (int st = 64; st > 0; st >>= 1) {
        if (tid < st) red[tid] += red[tid + st];
        __syncthreads();
    }
    float mu = red[0] / (float)CF;
    __syncthreads();
    float d0 = x0 - mu, d1 = x1 - mu, d2 = x2 - mu;
    red[tid] = d0 * d0 + d1 * d1 + d2 * d2;
    __syncthreads();
    for (int st = 64; st > 0; st >>= 1) {
        if (tid < st) red[tid] += red[tid + st];
        __syncthreads();
    }
    float rs = rsqrtf(red[0] / (float)CF + 1e-5f);
    srow[tid]       = d0 * rs * g[tid]       + b[tid];
    srow[tid + 128] = d1 * rs * g[tid + 128] + b[tid + 128];
    srow[tid + 256] = d2 * rs * g[tid + 256] + b[tid + 256];
    __syncthreads();
    int c0 = tid, c1 = tid + 128;
    float al0 = 0.f, al1 = 0.f, am0 = 0.f, am1 = 0.f;
    for (int f = 0; f < CF; ++f) {
        float sv = srow[f];
        al0 += sv * Wl[f * CP + c0];
        al1 += sv * Wl[f * CP + c1];
        am0 += sv * Wm[f * CP + c0];
        am1 += sv * Wm[f * CP + c1];
    }
    lo[n * CP + c0] = al0;
    lo[n * CP + c1] = al1;
    mo[n * CP + c0] = am0;
    mo[n * CP + c1] = am1;
}

// ---------- fused main kernel ----------
__global__ __launch_bounds__(256)
void fused_kernel(const float* __restrict__ z, const float* __restrict__ fpe,
                  const long long* __restrict__ ridx, const float* __restrict__ rmask,
                  const float* __restrict__ zlg, const float* __restrict__ zlb,
                  const float* __restrict__ ffg, const float* __restrict__ ffb,
                  const float* __restrict__ bs1, const float* __restrict__ bs2,
                  const float* __restrict__ bs3,
                  const float* __restrict__ lbuf, const float* __restrict__ mbuf,
                  const unsigned short* __restrict__ wzp,
                  const unsigned short* __restrict__ w1p,
                  const unsigned short* __restrict__ w2p,
                  const unsigned short* __restrict__ w3p,
                  float* __restrict__ out) {
    __shared__ __align__(16) float  feS[TM * CP];       // 16 KB
    __shared__ __align__(32) __bf16 apack[8 * 512];     // 8 KB (up to K=256)
    __shared__ float redS[256];
    __shared__ float redQ[256];

    const int tid = threadIdx.x;
    const int row = tid >> 4, sub = tid & 15;
    const long long pbase = (long long)blockIdx.x * TM;
    const long long prow  = pbase + row;
    const int irow = (int)(prow >> 9), jrow = (int)(prow & 511);

    // ===== Phase 0: feS = framepair + l_i + m_j (b128 NT loads, coalesced) =====
    {
        const f4* fp4 = (const f4*)(fpe + (size_t)prow * CP + sub * 16);
        const f4* l4  = (const f4*)(lbuf + (size_t)irow * CP + sub * 16);
        const f4* m4  = (const f4*)(mbuf + (size_t)jrow * CP + sub * 16);
        f4* fs4 = (f4*)(feS + row * CP + sub * 16);
#pragma unroll
        for (int v = 0; v < 4; ++v) {
            f4 a = __builtin_nontemporal_load(fp4 + v);   // streamed once: keep L2 for weights
            fs4[v] = a + l4[v] + m4[v];
        }
    }

    // ===== Phase A: gather z rows, mask, LayerNorm, pack bf16 A (K=128) =====
    {
        long long zi = ridx[irow], zj = ridx[jrow];
        float mk = rmask[irow] * rmask[jrow];
        const f4* zr4 = (const f4*)(z + ((size_t)zi * NRES + (size_t)zj) * CZ + sub * 8);
        f4 zv0 = zr4[0], zv1 = zr4[1];
        float xv[8];
        float s = 0.f, q = 0.f;
#pragma unroll
        for (int e2 = 0; e2 < 8; ++e2) {
            float v = (e2 < 4 ? zv0[e2] : zv1[e2 - 4]) * mk;
            xv[e2] = v;
            s += v;
            q += v * v;
        }
        redS[tid] = s; redQ[tid] = q;
        __syncthreads();
        for (int st = 8; st > 0; st >>= 1) {
            if (sub < st) { redS[tid] += redS[tid + st]; redQ[tid] += redQ[tid + st]; }
            __syncthreads();
        }
        float mu  = redS[row << 4] / (float)CZ;
        float var = redQ[row << 4] / (float)CZ - mu * mu;
        float rs  = rsqrtf(var + 1e-5f);
#pragma unroll
        for (int e2 = 0; e2 < 8; ++e2) {
            int k = sub * 8 + e2;
            float v = (xv[e2] - mu) * rs * zlg[k] + zlb[k];
            apack[(k >> 5) * 512 + pack_idx_a(row, k & 31)] = (__bf16)v;
        }
        __syncthreads();   // apack + feS complete
    }

    const int w     = tid >> 5, lane = tid & 31;
    const int nt0   = 2 * w, nt1 = nt0 + 1;
    const int ncol0 = nt0 * 16 + (lane & 15);
    const int ncol1 = ncol0 + 16;
    const int mhalf = (lane >> 4) * 8;

    // ===== GEMM1: LN(zb) @ W_z  (K=128, 4 k-tiles) =====
    v8f acc0 = vzero8(), acc1 = vzero8();
#pragma unroll
    for (int kt = 0; kt < 4; ++kt) {
        v16bf a  = *(const v16bf*)(apack + kt * 512 + lane * 16);
        v16bf bA = *(const v16bf*)((const __bf16*)wzp + ((size_t)(nt0 * 4 + kt) * 32 + lane) * 16);
        v16bf bB = *(const v16bf*)((const __bf16*)wzp + ((size_t)(nt1 * 4 + kt) * 32 + lane) * 16);
        acc0 = wmma_bf16(a, bA, acc0);
        acc1 = wmma_bf16(a, bB, acc1);
    }
    // fe += zW_z (RMW into LDS tile; waves own disjoint column slices)
#pragma unroll
    for (int r = 0; r < 8; ++r) {
        int mrow = r + mhalf;
        feS[mrow * CP + ncol0] += acc0[r];
        feS[mrow * CP + ncol1] += acc1[r];
    }
    __syncthreads();

    // ===== Phase B: LN(fe) -> A pack (K=256, 8 k-tiles) =====
    {
        float xv[16];
        float s = 0.f, q = 0.f;
#pragma unroll
        for (int e2 = 0; e2 < 16; ++e2) {
            float v = feS[row * CP + sub * 16 + e2];
            xv[e2] = v;
            s += v;
            q += v * v;
        }
        redS[tid] = s; redQ[tid] = q;
        __syncthreads();
        for (int st = 8; st > 0; st >>= 1) {
            if (sub < st) { redS[tid] += redS[tid + st]; redQ[tid] += redQ[tid + st]; }
            __syncthreads();
        }
        float mu  = redS[row << 4] / (float)CP;
        float var = redQ[row << 4] / (float)CP - mu * mu;
        float rs  = rsqrtf(var + 1e-5f);
#pragma unroll
        for (int e2 = 0; e2 < 16; ++e2) {
            int k = sub * 16 + e2;
            float v = (xv[e2] - mu) * rs * ffg[k] + ffb[k];
            apack[(k >> 5) * 512 + pack_idx_a(row, k & 31)] = (__bf16)v;
        }
        __syncthreads();
    }

    // ===== FFN layer 1: relu(h @ W1 + b1) =====
    acc0 = vzero8(); acc1 = vzero8();
#pragma unroll
    for (int kt = 0; kt < 8; ++kt) {
        v16bf a  = *(const v16bf*)(apack + kt * 512 + lane * 16);
        v16bf bA = *(const v16bf*)((const __bf16*)w1p + ((size_t)(nt0 * 8 + kt) * 32 + lane) * 16);
        v16bf bB = *(const v16bf*)((const __bf16*)w1p + ((size_t)(nt1 * 8 + kt) * 32 + lane) * 16);
        acc0 = wmma_bf16(a, bA, acc0);
        acc1 = wmma_bf16(a, bB, acc1);
    }
    __syncthreads();   // all waves done reading apack
    {
        float bc0 = bs1[ncol0], bc1 = bs1[ncol1];
#pragma unroll
        for (int r = 0; r < 8; ++r) {
            int mrow = r + mhalf;
            apack[(ncol0 >> 5) * 512 + pack_idx_a(mrow, ncol0 & 31)] = (__bf16)fmaxf(acc0[r] + bc0, 0.f);
            apack[(ncol1 >> 5) * 512 + pack_idx_a(mrow, ncol1 & 31)] = (__bf16)fmaxf(acc1[r] + bc1, 0.f);
        }
    }
    __syncthreads();

    // ===== FFN layer 2: relu(h @ W2 + b2) =====
    acc0 = vzero8(); acc1 = vzero8();
#pragma unroll
    for (int kt = 0; kt < 8; ++kt) {
        v16bf a  = *(const v16bf*)(apack + kt * 512 + lane * 16);
        v16bf bA = *(const v16bf*)((const __bf16*)w2p + ((size_t)(nt0 * 8 + kt) * 32 + lane) * 16);
        v16bf bB = *(const v16bf*)((const __bf16*)w2p + ((size_t)(nt1 * 8 + kt) * 32 + lane) * 16);
        acc0 = wmma_bf16(a, bA, acc0);
        acc1 = wmma_bf16(a, bB, acc1);
    }
    __syncthreads();
    {
        float bc0 = bs2[ncol0], bc1 = bs2[ncol1];
#pragma unroll
        for (int r = 0; r < 8; ++r) {
            int mrow = r + mhalf;
            apack[(ncol0 >> 5) * 512 + pack_idx_a(mrow, ncol0 & 31)] = (__bf16)fmaxf(acc0[r] + bc0, 0.f);
            apack[(ncol1 >> 5) * 512 + pack_idx_a(mrow, ncol1 & 31)] = (__bf16)fmaxf(acc1[r] + bc1, 0.f);
        }
    }
    __syncthreads();

    // ===== FFN layer 3 + residual: fe += h @ W3 + b3 (into LDS) =====
    acc0 = vzero8(); acc1 = vzero8();
#pragma unroll
    for (int kt = 0; kt < 8; ++kt) {
        v16bf a  = *(const v16bf*)(apack + kt * 512 + lane * 16);
        v16bf bA = *(const v16bf*)((const __bf16*)w3p + ((size_t)(nt0 * 8 + kt) * 32 + lane) * 16);
        v16bf bB = *(const v16bf*)((const __bf16*)w3p + ((size_t)(nt1 * 8 + kt) * 32 + lane) * 16);
        acc0 = wmma_bf16(a, bA, acc0);
        acc1 = wmma_bf16(a, bB, acc1);
    }
    {
        float bc0 = bs3[ncol0], bc1 = bs3[ncol1];
#pragma unroll
        for (int r = 0; r < 8; ++r) {
            int mrow = r + mhalf;
            feS[mrow * CP + ncol0] += acc0[r] + bc0;
            feS[mrow * CP + ncol1] += acc1[r] + bc1;
        }
    }
    __syncthreads();

    // ===== cooperative b128 NT store of the finished tile =====
    {
        const f4* fs4 = (const f4*)(feS + row * CP + sub * 16);
        f4* o4 = (f4*)(out + (size_t)prow * CP + sub * 16);
#pragma unroll
        for (int v = 0; v < 4; ++v)
            __builtin_nontemporal_store(fs4[v], o4 + v);
    }
}

extern "C" void kernel_launch(void* const* d_in, const int* in_sizes, int n_in,
                              void* d_out, int out_size, void* d_ws, size_t ws_size,
                              hipStream_t stream) {
    const float*     z     = (const float*)d_in[0];
    const float*     fpe   = (const float*)d_in[1];
    const float*     rig   = (const float*)d_in[2];
    const long long* ridx  = (const long long*)d_in[3];
    const float*     rmask = (const float*)d_in[4];
    const float*     zlg   = (const float*)d_in[5];
    const float*     zlb   = (const float*)d_in[6];
    const float*     Wz    = (const float*)d_in[7];
    const float*     flg   = (const float*)d_in[8];
    const float*     flb   = (const float*)d_in[9];
    const float*     Wl    = (const float*)d_in[10];
    const float*     Wm    = (const float*)d_in[11];
    const float*     ffg   = (const float*)d_in[12];
    const float*     ffb   = (const float*)d_in[13];
    const float*     W1    = (const float*)d_in[14];
    const float*     B1    = (const float*)d_in[15];
    const float*     W2    = (const float*)d_in[16];
    const float*     B2    = (const float*)d_in[17];
    const float*     W3    = (const float*)d_in[18];
    const float*     B3    = (const float*)d_in[19];
    float* out = (float*)d_out;

    // workspace layout
    float* lbuf = (float*)d_ws;                       // 512*256 f32
    float* mbuf = lbuf + NRES * CP;                   // 512*256 f32
    unsigned short* wzp = (unsigned short*)(mbuf + NRES * CP); // 16*4*512 bf16
    unsigned short* w1p = wzp + 16 * 4 * 512;         // 16*8*512 bf16 each
    unsigned short* w2p = w1p + 16 * 8 * 512;
    unsigned short* w3p = w2p + 16 * 8 * 512;

    prepack_kernel<<<(16 * 4 * 512) / 256, 256, 0, stream>>>(Wz, wzp, 4);
    prepack_kernel<<<(16 * 8 * 512) / 256, 256, 0, stream>>>(W1, w1p, 8);
    prepack_kernel<<<(16 * 8 * 512) / 256, 256, 0, stream>>>(W2, w2p, 8);
    prepack_kernel<<<(16 * 8 * 512) / 256, 256, 0, stream>>>(W3, w3p, 8);
    lm_kernel<<<NRES, 128, 0, stream>>>(rig, flg, flb, Wl, Wm, lbuf, mbuf);
    fused_kernel<<<(NRES * NRES) / TM, 256, 0, stream>>>(
        z, fpe, ridx, rmask, zlg, zlb, ffg, ffb, B1, B2, B3,
        lbuf, mbuf, wzp, w1p, w2p, w3p, out);
}
